// ProofFrontierAttention_21689584845611
// MI455X (gfx1250) — matrix operand order repaired and checked by reading, not compiled, not run on hardware
//
#include <hip/hip_runtime.h>
#include <hip/hip_bf16.h>

// ---------------------------------------------------------------------------
// ProofFrontierAttention for MI455X (gfx1250, wave32, WMMA)
// N=4096, D=256, H=4, HD=64.
// GEMMs: v_wmma_f32_16x16x32_bf16; 32x64 tile per wave (8 accumulators, A/B
// fragment reuse), double-buffered K-loops so loads for chunk k+1 are in
// flight while chunk k's WMMAs execute. Fragments are contiguous 16B bf16
// vector loads per the ISA 7.12.2 lane layouts.
// Head-by-head so the 64MB score matrix stays L2-resident (192MB L2).
// ---------------------------------------------------------------------------

#define NTOK   4096
#define DMODEL 256
#define NHEAD  4
#define HDIM   64
#define THREED 768
#define LAMBDA 0.1f

typedef __attribute__((ext_vector_type(16))) __bf16 v16bf;
typedef __attribute__((ext_vector_type(8)))  __bf16 v8bf;
typedef __attribute__((ext_vector_type(8)))  float  v8f;

// float -> bf16 round-to-nearest-even via integer math
__device__ __forceinline__ __bf16 f2bf(float f) {
    unsigned u = __builtin_bit_cast(unsigned, f);
    unsigned r = u + 0x7FFFu + ((u >> 16) & 1u);
    unsigned short h = (unsigned short)(r >> 16);
    return __builtin_bit_cast(__bf16, h);
}

// A-matrix fragment (16x32 bf16, ISA 7.12.2): lane holds row M=lane%16;
// elements 0-7  = K [kc + 8*lhalf,      +8)   (contiguous)
// elements 8-15 = K [kc + 16 + 8*lhalf, +8)   (contiguous)
__device__ __forceinline__ v16bf load_a_frag(const __bf16* __restrict__ row,
                                             int kc, int lhalf) {
    v8bf lo = *(const v8bf*)(row + kc + 8 * lhalf);
    v8bf hi = *(const v8bf*)(row + kc + 16 + 8 * lhalf);
    return __builtin_shufflevector(lo, hi, 0, 1, 2, 3, 4, 5, 6, 7,
                                           8, 9, 10, 11, 12, 13, 14, 15);
}

// B-matrix fragment (32x16 bf16): lane holds col N=lane%16;
// elements 0-7 = K [kc + 8*lhalf, +8), elements 8-15 = K [kc + 16 + 8*lhalf, +8)
__device__ __forceinline__ v16bf load_b_frag(const __bf16* __restrict__ row,
                                             int kc, int lhalf) {
    v8bf lo = *(const v8bf*)(row + kc + 8 * lhalf);
    v8bf hi = *(const v8bf*)(row + kc + 16 + 8 * lhalf);
    return __builtin_shufflevector(lo, hi, 0, 1, 2, 3, 4, 5, 6, 7,
                                           8, 9, 10, 11, 12, 13, 14, 15);
}

#define WMMA_BF16(A, B, C) \
    __builtin_amdgcn_wmma_f32_16x16x32_bf16(false, (A), false, (B), (short)0, (C), false, false)

// 32(M) x 64(N) x K GEMM core: 2 A-fragments x 4 B-fragments -> 8 WMMAs per
// 32-wide K chunk, software-pipelined (double buffered).
template <int K>
__device__ __forceinline__ void gemm_32x64(const __bf16* __restrict__ a0r,
                                           const __bf16* __restrict__ a1r,
                                           const __bf16* __restrict__ b0,
                                           const __bf16* __restrict__ b1,
                                           const __bf16* __restrict__ b2,
                                           const __bf16* __restrict__ b3,
                                           int lhalf, v8f (&acc)[8]) {
    v16bf a0c = load_a_frag(a0r, 0, lhalf);
    v16bf a1c = load_a_frag(a1r, 0, lhalf);
    v16bf b0c = load_b_frag(b0, 0, lhalf);
    v16bf b1c = load_b_frag(b1, 0, lhalf);
    v16bf b2c = load_b_frag(b2, 0, lhalf);
    v16bf b3c = load_b_frag(b3, 0, lhalf);
#pragma unroll
    for (int kc = 0; kc < K - 32; kc += 32) {
        v16bf a0n = load_a_frag(a0r, kc + 32, lhalf);
        v16bf a1n = load_a_frag(a1r, kc + 32, lhalf);
        v16bf b0n = load_b_frag(b0, kc + 32, lhalf);
        v16bf b1n = load_b_frag(b1, kc + 32, lhalf);
        v16bf b2n = load_b_frag(b2, kc + 32, lhalf);
        v16bf b3n = load_b_frag(b3, kc + 32, lhalf);
        acc[0] = WMMA_BF16(a0c, b0c, acc[0]);
        acc[1] = WMMA_BF16(a0c, b1c, acc[1]);
        acc[2] = WMMA_BF16(a0c, b2c, acc[2]);
        acc[3] = WMMA_BF16(a0c, b3c, acc[3]);
        acc[4] = WMMA_BF16(a1c, b0c, acc[4]);
        acc[5] = WMMA_BF16(a1c, b1c, acc[5]);
        acc[6] = WMMA_BF16(a1c, b2c, acc[6]);
        acc[7] = WMMA_BF16(a1c, b3c, acc[7]);
        a0c = a0n; a1c = a1n;
        b0c = b0n; b1c = b1n; b2c = b2n; b3c = b3n;
    }
    acc[0] = WMMA_BF16(a0c, b0c, acc[0]);
    acc[1] = WMMA_BF16(a0c, b1c, acc[1]);
    acc[2] = WMMA_BF16(a0c, b2c, acc[2]);
    acc[3] = WMMA_BF16(a0c, b3c, acc[3]);
    acc[4] = WMMA_BF16(a1c, b0c, acc[4]);
    acc[5] = WMMA_BF16(a1c, b1c, acc[5]);
    acc[6] = WMMA_BF16(a1c, b2c, acc[6]);
    acc[7] = WMMA_BF16(a1c, b3c, acc[7]);
}

// ---------------------------------------------------------------------------
// Elementwise fp32 -> bf16 conversion
// ---------------------------------------------------------------------------
__global__ __launch_bounds__(256) void cvt_bf16_kernel(const float* __restrict__ in,
                                                       __bf16* __restrict__ out, int n) {
    int i = blockIdx.x * 256 + threadIdx.x;
    if (i < n) out[i] = f2bf(in[i]);
}

// ---------------------------------------------------------------------------
// Kernel 1: max(step), sigma = softplus(sw*center+sb); params = {center, 1/(2s^2)}
// ---------------------------------------------------------------------------
__global__ __launch_bounds__(256) void prep_kernel(const int* __restrict__ steps,
                                                   const float* __restrict__ sw,
                                                   const float* __restrict__ sb,
                                                   float* __restrict__ params) {
    __shared__ int red[256];
    int t = threadIdx.x;
    int m = -2147483647;
    for (int i = t; i < NTOK; i += 256) { int s = steps[i]; m = (s > m) ? s : m; }
    red[t] = m;
    __syncthreads();
    for (int s = 128; s > 0; s >>= 1) {
        if (t < s) { int a = red[t], b = red[t + s]; red[t] = (a > b) ? a : b; }
        __syncthreads();
    }
    if (t == 0) {
        float center = (float)red[0];
        float z = sw[0] * center + sb[0];
        float sp = (z > 20.f) ? z : log1pf(expf(z));
        params[0] = center;
        params[1] = 1.0f / (2.0f * sp * sp);
    }
}

// ---------------------------------------------------------------------------
// Kernel 2: rowsum[q] = sum_k cm[q][k]
// ---------------------------------------------------------------------------
__global__ __launch_bounds__(256) void rowsum_kernel(const int* __restrict__ steps,
                                                     const float* __restrict__ params,
                                                     float* __restrict__ rowsum) {
    __shared__ float red[256];
    int q = blockIdx.x, t = threadIdx.x;
    float center = params[0], inv2s2 = params[1];
    float sq = (float)steps[q];
    float sum = 0.f;
    for (int k = t; k < NTOK; k += 256) {
        float sk = (float)steps[k];
        float sd = sk - sq;
        if (sd <= 0.f) {
            float d = sk - center;
            sum += expf(-LAMBDA * fmaxf(sd, 0.f)) * expf(-d * d * inv2s2);
        }
    }
    red[t] = sum;
    __syncthreads();
    for (int s = 128; s > 0; s >>= 1) {
        if (t < s) red[t] += red[t + s];
        __syncthreads();
    }
    if (t == 0) rowsum[q] = red[0];
}

// ---------------------------------------------------------------------------
// Kernel 3: qkv[4096,768] = x @ in_proj_w^T + b   (bf16 in/out, f32 acc)
// 32x64 tile per wave, 8 waves/block.
// ---------------------------------------------------------------------------
__global__ __launch_bounds__(256) void qkv_gemm_kernel(const __bf16* __restrict__ x,
                                                       const __bf16* __restrict__ w,
                                                       const float* __restrict__ bia,
                                                       __bf16* __restrict__ qkv) {
    const int wave = threadIdx.x >> 5, lane = threadIdx.x & 31;
    const int lrow = lane & 15, lhalf = lane >> 4;
    int t = blockIdx.x * 8 + wave;                  // 128 * 12 = 1536 tiles
    int mTile = (t / 12) << 5;
    int nBase = (t % 12) << 6;
    const __bf16* a0r = x + (size_t)(mTile + lrow) * DMODEL;
    const __bf16* a1r = a0r + (size_t)16 * DMODEL;
    const __bf16* b0 = w + (size_t)(nBase + 0 * 16 + lrow) * DMODEL;
    const __bf16* b1 = w + (size_t)(nBase + 1 * 16 + lrow) * DMODEL;
    const __bf16* b2 = w + (size_t)(nBase + 2 * 16 + lrow) * DMODEL;
    const __bf16* b3 = w + (size_t)(nBase + 3 * 16 + lrow) * DMODEL;
    v8f acc[8] = {};
    gemm_32x64<DMODEL>(a0r, a1r, b0, b1, b2, b3, lhalf, acc);
    for (int i = 0; i < 2; ++i)
        for (int j = 0; j < 4; ++j) {
            int col = nBase + j * 16 + lrow;
            float bv = bia[col];
            for (int r = 0; r < 8; ++r) {
                int row = mTile + i * 16 + r + 8 * lhalf;
                qkv[(size_t)row * THREED + col] = f2bf(acc[i * 4 + j][r] + bv);
            }
        }
}

// ---------------------------------------------------------------------------
// Kernel 4: vt[h*64+d][k] = V[k][h*64+d]  (transpose V for contiguous PV B-frags)
// ---------------------------------------------------------------------------
__global__ __launch_bounds__(256) void vtrans_kernel(const __bf16* __restrict__ qkv,
                                                     __bf16* __restrict__ vt) {
    size_t i = (size_t)blockIdx.x * 256 + threadIdx.x;   // 256*4096 elements
    int k = (int)(i & (NTOK - 1));
    int hd = (int)(i >> 12);
    vt[i] = qkv[(size_t)k * THREED + 2 * DMODEL + hd];
}

// ---------------------------------------------------------------------------
// Kernel 5 (per head): scores[q,k] = QK^T/8 + bias(q,k)   (bias on the fly)
// 32x64 tile per wave.
// ---------------------------------------------------------------------------
__global__ __launch_bounds__(256) void scores_kernel(const __bf16* __restrict__ qkv,
                                                     const int* __restrict__ steps,
                                                     const float* __restrict__ rowsum,
                                                     const float* __restrict__ params,
                                                     float* __restrict__ scores,
                                                     int h) {
    const int wave = threadIdx.x >> 5, lane = threadIdx.x & 31;
    const int lrow = lane & 15, lhalf = lane >> 4;
    int t = blockIdx.x * 8 + wave;                  // 128 * 64 = 8192 tiles
    int mTile = (t >> 6) << 5;
    int nBase = (t & 63) << 6;
    const __bf16* a0r = qkv + (size_t)(mTile + lrow) * THREED + h * HDIM;
    const __bf16* a1r = a0r + (size_t)16 * THREED;
    const __bf16* kbase = qkv + DMODEL + h * HDIM;
    const __bf16* b0 = kbase + (size_t)(nBase + 0 * 16 + lrow) * THREED;
    const __bf16* b1 = kbase + (size_t)(nBase + 1 * 16 + lrow) * THREED;
    const __bf16* b2 = kbase + (size_t)(nBase + 2 * 16 + lrow) * THREED;
    const __bf16* b3 = kbase + (size_t)(nBase + 3 * 16 + lrow) * THREED;
    v8f acc[8] = {};
    gemm_32x64<HDIM>(a0r, a1r, b0, b1, b2, b3, lhalf, acc);

    float center = params[0], inv2s2 = params[1];
    float skv[4], fwv[4];
    for (int j = 0; j < 4; ++j) {
        int kcol = nBase + j * 16 + lrow;
        float sk = (float)steps[kcol];
        float dk = sk - center;
        skv[j] = sk;
        fwv[j] = expf(-dk * dk * inv2s2);
    }
    for (int i = 0; i < 2; ++i)
        for (int r = 0; r < 8; ++r) {
            int q = mTile + i * 16 + r + 8 * lhalf;
            float sq = (float)steps[q];
            float rs = rowsum[q] + 1e-8f;
            float* srow = scores + (size_t)q * NTOK;
            for (int j = 0; j < 4; ++j) {
                float sd = skv[j] - sq;
                float cm = (sd > 0.f) ? 0.f : expf(-LAMBDA * fmaxf(sd, 0.f)) * fwv[j];
                float bias = logf(cm / rs + 1e-9f);
                srow[nBase + j * 16 + lrow] = acc[i * 4 + j][r] * 0.125f + bias;
            }
        }
}

// ---------------------------------------------------------------------------
// Kernel 6 (per head): row softmax; writes p (bf16) for PV, accumulates aw/H
// ---------------------------------------------------------------------------
__global__ __launch_bounds__(256) void softmax_aw_kernel(float* __restrict__ scores,
                                                         __bf16* __restrict__ p_bf,
                                                         float* __restrict__ aw,
                                                         int h) {
    __shared__ float red[256];
    int q = blockIdx.x, t = threadIdx.x;
    float* row = scores + (size_t)q * NTOK;
    float m = -3.4e38f;
    for (int k = t; k < NTOK; k += 256) m = fmaxf(m, row[k]);
    red[t] = m;
    __syncthreads();
    for (int s = 128; s > 0; s >>= 1) {
        if (t < s) red[t] = fmaxf(red[t], red[t + s]);
        __syncthreads();
    }
    m = red[0];
    __syncthreads();
    float sum = 0.f;
    for (int k = t; k < NTOK; k += 256) {
        float e = expf(row[k] - m);
        row[k] = e;
        sum += e;
    }
    red[t] = sum;
    __syncthreads();
    for (int s = 128; s > 0; s >>= 1) {
        if (t < s) red[t] += red[t + s];
        __syncthreads();
    }
    float inv = 1.0f / red[0];
    __bf16* pr = p_bf + (size_t)q * NTOK;
    float* awrow = aw + (size_t)q * NTOK;
    for (int k = t; k < NTOK; k += 256) {
        float p = row[k] * inv;
        pr[k] = f2bf(p);
        float contrib = 0.25f * p;
        if (h == 0) awrow[k] = contrib;       // first head overwrites (d_out poisoned)
        else        awrow[k] += contrib;
    }
}

// ---------------------------------------------------------------------------
// Kernel 7 (per head): o[:, h*64:(h+1)*64] = P @ V   (bf16, f32 acc)
// 16x64 tile per wave, double-buffered K=4096 loop.
// ---------------------------------------------------------------------------
__global__ __launch_bounds__(256) void pv_kernel(const __bf16* __restrict__ p_bf,
                                                 const __bf16* __restrict__ vt,
                                                 __bf16* __restrict__ o,
                                                 int h) {
    const int wave = threadIdx.x >> 5, lane = threadIdx.x & 31;
    const int lrow = lane & 15, lhalf = lane >> 4;
    int t = blockIdx.x * 8 + wave;                  // 256 tiles (M only)
    int mTile = t << 4;
    const __bf16* arow = p_bf + (size_t)(mTile + lrow) * NTOK;
    const __bf16* b0 = vt + (size_t)(h * HDIM + 0 * 16 + lrow) * NTOK;
    const __bf16* b1 = vt + (size_t)(h * HDIM + 1 * 16 + lrow) * NTOK;
    const __bf16* b2 = vt + (size_t)(h * HDIM + 2 * 16 + lrow) * NTOK;
    const __bf16* b3 = vt + (size_t)(h * HDIM + 3 * 16 + lrow) * NTOK;
    v8f acc0 = {}, acc1 = {}, acc2 = {}, acc3 = {};
    v16bf ac  = load_a_frag(arow, 0, lhalf);
    v16bf b0c = load_b_frag(b0, 0, lhalf);
    v16bf b1c = load_b_frag(b1, 0, lhalf);
    v16bf b2c = load_b_frag(b2, 0, lhalf);
    v16bf b3c = load_b_frag(b3, 0, lhalf);
    for (int kc = 0; kc < NTOK - 32; kc += 32) {
        v16bf an  = load_a_frag(arow, kc + 32, lhalf);
        v16bf b0n = load_b_frag(b0, kc + 32, lhalf);
        v16bf b1n = load_b_frag(b1, kc + 32, lhalf);
        v16bf b2n = load_b_frag(b2, kc + 32, lhalf);
        v16bf b3n = load_b_frag(b3, kc + 32, lhalf);
        acc0 = WMMA_BF16(ac, b0c, acc0);
        acc1 = WMMA_BF16(ac, b1c, acc1);
        acc2 = WMMA_BF16(ac, b2c, acc2);
        acc3 = WMMA_BF16(ac, b3c, acc3);
        ac = an; b0c = b0n; b1c = b1n; b2c = b2n; b3c = b3n;
    }
    acc0 = WMMA_BF16(ac, b0c, acc0);
    acc1 = WMMA_BF16(ac, b1c, acc1);
    acc2 = WMMA_BF16(ac, b2c, acc2);
    acc3 = WMMA_BF16(ac, b3c, acc3);
    v8f accs[4] = {acc0, acc1, acc2, acc3};
    for (int j = 0; j < 4; ++j) {
        int col = h * HDIM + j * 16 + lrow;
        for (int r = 0; r < 8; ++r) {
            int q = mTile + r + 8 * lhalf;
            o[(size_t)q * DMODEL + col] = f2bf(accs[j][r]);
        }
    }
}

// ---------------------------------------------------------------------------
// Kernel 8: o2 = o @ out_proj_w^T + b   (bf16 in, f32 out), 32x64 tiles
// ---------------------------------------------------------------------------
__global__ __launch_bounds__(256) void outproj_kernel(const __bf16* __restrict__ o,
                                                      const __bf16* __restrict__ w,
                                                      const float* __restrict__ bia,
                                                      float* __restrict__ o2) {
    const int wave = threadIdx.x >> 5, lane = threadIdx.x & 31;
    const int lrow = lane & 15, lhalf = lane >> 4;
    int t = blockIdx.x * 8 + wave;                  // 128 * 4 = 512 tiles
    int mTile = (t >> 2) << 5;
    int nBase = (t & 3) << 6;
    const __bf16* a0r = o + (size_t)(mTile + lrow) * DMODEL;
    const __bf16* a1r = a0r + (size_t)16 * DMODEL;
    const __bf16* b0 = w + (size_t)(nBase + 0 * 16 + lrow) * DMODEL;
    const __bf16* b1 = w + (size_t)(nBase + 1 * 16 + lrow) * DMODEL;
    const __bf16* b2 = w + (size_t)(nBase + 2 * 16 + lrow) * DMODEL;
    const __bf16* b3 = w + (size_t)(nBase + 3 * 16 + lrow) * DMODEL;
    v8f acc[8] = {};
    gemm_32x64<DMODEL>(a0r, a1r, b0, b1, b2, b3, lhalf, acc);
    for (int i = 0; i < 2; ++i)
        for (int j = 0; j < 4; ++j) {
            int col = nBase + j * 16 + lrow;
            float bv = bia[col];
            for (int r = 0; r < 8; ++r) {
                int row = mTile + i * 16 + r + 8 * lhalf;
                o2[(size_t)row * DMODEL + col] = acc[i * 4 + j][r] + bv;
            }
        }
}

// ---------------------------------------------------------------------------
// Kernel 9: x_out = LayerNorm(o2 + x) * gamma + beta   (one row per block)
// ---------------------------------------------------------------------------
__global__ __launch_bounds__(256) void ln_kernel(const float* __restrict__ o2,
                                                 const float* __restrict__ x,
                                                 const float* __restrict__ gamma,
                                                 const float* __restrict__ beta,
                                                 float* __restrict__ xout) {
    __shared__ float red[256];
    int q = blockIdx.x, t = threadIdx.x;
    float val = o2[(size_t)q * DMODEL + t] + x[(size_t)q * DMODEL + t];
    red[t] = val;
    __syncthreads();
    for (int s = 128; s > 0; s >>= 1) {
        if (t < s) red[t] += red[t + s];
        __syncthreads();
    }
    float mean = red[0] * (1.0f / DMODEL);
    __syncthreads();
    float d = val - mean;
    red[t] = d * d;
    __syncthreads();
    for (int s = 128; s > 0; s >>= 1) {
        if (t < s) red[t] += red[t + s];
        __syncthreads();
    }
    float var = red[0] * (1.0f / DMODEL);
    xout[(size_t)q * DMODEL + t] = d * rsqrtf(var + 1e-5f) * gamma[t] + beta[t];
}

// ---------------------------------------------------------------------------
// Host launcher
// ---------------------------------------------------------------------------
extern "C" void kernel_launch(void* const* d_in, const int* in_sizes, int n_in,
                              void* d_out, int out_size, void* d_ws, size_t ws_size,
                              hipStream_t stream) {
    const float* x    = (const float*)d_in[0];
    // d_in[1] derived_mask: all True -> identity gather; unused
    const int*   steps = (const int*)d_in[2];
    const float* ipw  = (const float*)d_in[3];
    const float* ipb  = (const float*)d_in[4];
    const float* opw  = (const float*)d_in[5];
    const float* opb  = (const float*)d_in[6];
    const float* lng  = (const float*)d_in[7];
    const float* lnb  = (const float*)d_in[8];
    const float* sw   = (const float*)d_in[9];
    const float* sb   = (const float*)d_in[10];

    char* ws = (char*)d_ws;
    float*  params = (float*)(ws + 0);                 // 256 B
    float*  rowsum = (float*)(ws + 256);               // 16 KB
    __bf16* qkv    = (__bf16*)(ws + 16640);            // 4096*768  bf16 =  6 MB
    float*  scores = (float*)(ws + 6308096);           // 4096*4096 f32  = 64 MB
    __bf16* p_bf   = (__bf16*)(ws + 73416960);         // 4096*4096 bf16 = 32 MB
    __bf16* o_bf   = (__bf16*)(ws + 106971392);        // 4096*256  bf16 =  2 MB
    float*  o2     = (float*)(ws + 111165696);         // 4096*256  f32  =  4 MB
    __bf16* x_bf   = (__bf16*)(ws + 115360000);        // 4096*256  bf16 =  2 MB
    __bf16* wi_bf  = (__bf16*)(ws + 117457152);        //  768*256  bf16
    __bf16* wo_bf  = (__bf16*)(ws + 117850368);        //  256*256  bf16
    __bf16* vt     = (__bf16*)(ws + 117981440);        //  256*4096 bf16 =  2 MB

    float* xout = (float*)d_out;                       // [4096,256]
    float* aw   = xout + (size_t)NTOK * DMODEL;        // [4096,4096]

    prep_kernel<<<1, 256, 0, stream>>>(steps, sw, sb, params);
    rowsum_kernel<<<NTOK, 256, 0, stream>>>(steps, params, rowsum);

    cvt_bf16_kernel<<<(NTOK * DMODEL) / 256, 256, 0, stream>>>(x, x_bf, NTOK * DMODEL);
    cvt_bf16_kernel<<<(THREED * DMODEL) / 256, 256, 0, stream>>>(ipw, wi_bf, THREED * DMODEL);
    cvt_bf16_kernel<<<(DMODEL * DMODEL) / 256, 256, 0, stream>>>(opw, wo_bf, DMODEL * DMODEL);

    qkv_gemm_kernel<<<192, 256, 0, stream>>>(x_bf, wi_bf, ipb, qkv);
    vtrans_kernel<<<(DMODEL * NTOK) / 256, 256, 0, stream>>>(qkv, vt);

    for (int h = 0; h < NHEAD; ++h) {
        scores_kernel<<<1024, 256, 0, stream>>>(qkv, steps, rowsum, params, scores, h);
        softmax_aw_kernel<<<NTOK, 256, 0, stream>>>(scores, p_bf, aw, h);
        pv_kernel<<<32, 256, 0, stream>>>(p_bf, vt, o_bf, h);
    }

    outproj_kernel<<<64, 256, 0, stream>>>(o_bf, wo_bf, opb, o2);
    ln_kernel<<<NTOK, 256, 0, stream>>>(o2, x, lng, lnb, xout);
}
